// GradientWeightedSparseAttention_74577812127907
// MI455X (gfx1250) — compile-verified, matrix operand order checked by reading, not run on hardware
//
#include <hip/hip_runtime.h>
#include <math.h>

typedef __bf16 bf16;
typedef __attribute__((ext_vector_type(16))) __bf16 v16bf;
typedef __attribute__((ext_vector_type(8)))  float  v8f;

#define BN 2
#define TT 2048
#define CC 1024
#define HH 16
#define DD 64

// Load a 16-element bf16 WMMA fragment: 8 contiguous bf16 from p0, 8 from p1.
// Matches the CDNA5 16-bit A/B layout: per-lane k-chunks of 8 at offsets
// 8*(lane>>4) and 16+8*(lane>>4).
static __device__ inline v16bf load_frag(const bf16* p0, const bf16* p1) {
  union U { uint4 u; bf16 h[8]; };
  U a, b;
  a.u = *(const uint4*)p0;
  b.u = *(const uint4*)p1;
  v16bf f;
#pragma unroll
  for (int i = 0; i < 8; ++i) { f[i] = a.h[i]; f[i + 8] = b.h[i]; }
  return f;
}

static __device__ inline v8f wmma_bf16(v16bf a, v16bf b, v8f c) {
  return __builtin_amdgcn_wmma_f32_16x16x32_bf16(false, a, false, b, (short)0, c,
                                                 false, false);
}

// ---------------- elementwise conversion / transpose ----------------

__global__ void cvt_f32_bf16(const float* __restrict__ src, bf16* __restrict__ dst, int n) {
  int i = blockIdx.x * blockDim.x + threadIdx.x;
  if (i < n) dst[i] = (bf16)src[i];
}

// src is [K][N] row-major f32; dst is [N][K] row-major bf16 (i.e. src^T).
__global__ void transpose_f32_bf16(const float* __restrict__ src, bf16* __restrict__ dst,
                                   int K, int N) {
  int o = blockIdx.x * blockDim.x + threadIdx.x;
  if (o >= K * N) return;
  int n = o / K;
  int k = o - n * K;
  dst[o] = (bf16)src[(size_t)k * N + n];
}

// ---------------- QKV GEMM: [4096,1024] x [1024,3072] ----------------
// 32x64 output tile per wave: 2 A-frags + 4 B-frags feed 8 WMMAs per k-step.
// WT is W_attn^T (bf16, [3072][1024]). Epilogue scatters into Q/K [B,H,T,D]
// (Q pre-scaled by 1/8) and Vt [B,H,D,T].

__global__ void qkv_gemm(const bf16* __restrict__ X, const bf16* __restrict__ WT,
                         const float* __restrict__ bias,
                         bf16* __restrict__ Q, bf16* __restrict__ K,
                         bf16* __restrict__ Vt) {
  const int lane = threadIdx.x & 31;
  const int wid  = threadIdx.x >> 5;
  const int tile = blockIdx.x * 4 + wid;
  const int NT = (3 * CC) / 64;           // 48 tiles along N
  const int tm = tile / NT;               // 0..127 (rows of 32)
  const int tn = tile - tm * NT;          // 0..47  (cols of 64)
  const int rlo = lane & 15, hi = lane >> 4;

  const bf16* arow0 = X + (size_t)(tm * 32 + rlo) * CC;
  const bf16* arow1 = X + (size_t)(tm * 32 + 16 + rlo) * CC;
  const bf16* brow[4];
#pragma unroll
  for (int ni = 0; ni < 4; ++ni)
    brow[ni] = WT + (size_t)(tn * 64 + ni * 16 + rlo) * CC;

  v8f acc[2][4] = {};
#pragma unroll 2
  for (int kk = 0; kk < CC; kk += 32) {
    const int kb = kk + hi * 8;
    v16bf a0 = load_frag(arow0 + kb, arow0 + kb + 16);
    v16bf a1 = load_frag(arow1 + kb, arow1 + kb + 16);
    v16bf b0 = load_frag(brow[0] + kb, brow[0] + kb + 16);
    v16bf b1 = load_frag(brow[1] + kb, brow[1] + kb + 16);
    v16bf b2 = load_frag(brow[2] + kb, brow[2] + kb + 16);
    v16bf b3 = load_frag(brow[3] + kb, brow[3] + kb + 16);
    acc[0][0] = wmma_bf16(a0, b0, acc[0][0]);
    acc[0][1] = wmma_bf16(a0, b1, acc[0][1]);
    acc[0][2] = wmma_bf16(a0, b2, acc[0][2]);
    acc[0][3] = wmma_bf16(a0, b3, acc[0][3]);
    acc[1][0] = wmma_bf16(a1, b0, acc[1][0]);
    acc[1][1] = wmma_bf16(a1, b1, acc[1][1]);
    acc[1][2] = wmma_bf16(a1, b2, acc[1][2]);
    acc[1][3] = wmma_bf16(a1, b3, acc[1][3]);
  }

#pragma unroll
  for (int ni = 0; ni < 4; ++ni) {
    const int j  = tn * 64 + ni * 16 + rlo;   // output column (N = lane&15)
    const float bj = bias[j];
#pragma unroll
    for (int mi = 0; mi < 2; ++mi) {
#pragma unroll
      for (int r = 0; r < 8; ++r) {
        const int m  = r + hi * 8;
        const int i  = tm * 32 + mi * 16 + m; // row in [0,4096)
        const int bb = i >> 11;               // batch
        const int t  = i & (TT - 1);
        float v = acc[mi][ni][r] + bj;
        if (j < CC) {                         // Q, fold softmax scale 1/sqrt(64)
          int hD = j >> 6, d = j & 63;
          Q[((size_t)(bb * HH + hD) * TT + t) * DD + d] = (bf16)(v * 0.125f);
        } else if (j < 2 * CC) {              // K
          int jj = j - CC, hD = jj >> 6, d = jj & 63;
          K[((size_t)(bb * HH + hD) * TT + t) * DD + d] = (bf16)v;
        } else {                              // V, transposed [B,H,D,T]
          int jj = j - 2 * CC, hD = jj >> 6, d = jj & 63;
          Vt[((size_t)(bb * HH + hD) * DD + d) * TT + t] = (bf16)v;
        }
      }
    }
  }
}

// ---------------- Flash-style causal attention ----------------
// One wave handles 32 queries (two 16-row tiles) for one (b,h). KV blocks of
// 32 keys: S via 8 WMMAs (K-frags shared across both q-tiles), online softmax
// in the C layout, P re-laid out through a 32x32 LDS tile into the A layout,
// then 8 WMMAs accumulate P@V (shared V-frags).

__global__ void attn(const bf16* __restrict__ Q, const bf16* __restrict__ K,
                     const bf16* __restrict__ Vt, bf16* __restrict__ Y) {
  __shared__ bf16 ldsP[4][32 * 32];
  const int lane = threadIdx.x & 31;
  const int wid  = threadIdx.x >> 5;
  bf16* P = ldsP[wid];

  const int w   = blockIdx.x * 4 + wid;   // 0..2047
  const int bh  = w >> 6;                 // b*16+h  (T/32 = 64 q-tiles)
  const int qt  = w & 63;
  const int qbase = qt * 32;
  const int rlo = lane & 15, hi = lane >> 4;

  const bf16* Qb = Q  + ((size_t)bh * TT + qbase) * DD;
  const bf16* Kb = K  + (size_t)bh * TT * DD;
  const bf16* Vb = Vt + (size_t)bh * DD * TT;

  v16bf qa[2][2];
#pragma unroll
  for (int qi = 0; qi < 2; ++qi)
#pragma unroll
    for (int kk = 0; kk < 2; ++kk) {
      const int db = kk * 32 + hi * 8;
      const bf16* p = Qb + (size_t)(qi * 16 + rlo) * DD + db;
      qa[qi][kk] = load_frag(p, p + 16);
    }

  v8f o[2][4] = {};
  float mR[2][8], lR[2][8];
#pragma unroll
  for (int qi = 0; qi < 2; ++qi)
#pragma unroll
    for (int r = 0; r < 8; ++r) { mR[qi][r] = -1e30f; lR[qi][r] = 0.f; }

  const int nblk = qt + 1;                // keys [0, qbase+32) in blocks of 32
  for (int kvb = 0; kvb < nblk; ++kvb) {
    const int kv0 = kvb * 32;

    // S = Q · K^T for 32 keys, both q-tiles (K-frags shared)
    v8f s[2][2] = {};
#pragma unroll
    for (int kk = 0; kk < 2; ++kk) {
      const int db = kk * 32 + hi * 8;
      const bf16* pk0 = Kb + (size_t)(kv0 + rlo) * DD + db;
      const bf16* pk1 = Kb + (size_t)(kv0 + 16 + rlo) * DD + db;
      v16bf b0 = load_frag(pk0, pk0 + 16);
      v16bf b1 = load_frag(pk1, pk1 + 16);
      s[0][0] = wmma_bf16(qa[0][kk], b0, s[0][0]);
      s[0][1] = wmma_bf16(qa[0][kk], b1, s[0][1]);
      s[1][0] = wmma_bf16(qa[1][kk], b0, s[1][0]);
      s[1][1] = wmma_bf16(qa[1][kk], b1, s[1][1]);
    }

    // causal mask + online softmax (row = r + 8*hi, col = rlo)
#pragma unroll
    for (int qi = 0; qi < 2; ++qi) {
#pragma unroll
      for (int r = 0; r < 8; ++r) {
        const int m  = r + hi * 8;
        const int qrow = qbase + qi * 16 + m;
        float a0 = (kv0 + rlo      > qrow) ? -__builtin_inff() : s[qi][0][r];
        float a1 = (kv0 + 16 + rlo > qrow) ? -__builtin_inff() : s[qi][1][r];
        float loc = fmaxf(a0, a1);
#pragma unroll
        for (int msk = 1; msk <= 8; msk <<= 1)
          loc = fmaxf(loc, __shfl_xor(loc, msk, 32));
        float mnew = fmaxf(mR[qi][r], loc);
        float corr = __expf(mR[qi][r] - mnew);
        float p0 = __expf(a0 - mnew);
        float p1 = __expf(a1 - mnew);
        float ls = p0 + p1;
#pragma unroll
        for (int msk = 1; msk <= 8; msk <<= 1)
          ls += __shfl_xor(ls, msk, 32);
        lR[qi][r] = lR[qi][r] * corr + ls;
        mR[qi][r] = mnew;
        o[qi][0][r] *= corr; o[qi][1][r] *= corr;
        o[qi][2][r] *= corr; o[qi][3][r] *= corr;
        P[(qi * 16 + m) * 32 + rlo]      = (bf16)p0;
        P[(qi * 16 + m) * 32 + 16 + rlo] = (bf16)p1;
      }
    }

    // P (32x32, A layout from LDS) @ V (32x64 via Vt rows), V-frags shared
    v16bf pa0 = load_frag(P + (0 * 16 + rlo) * 32 + hi * 8,
                          P + (0 * 16 + rlo) * 32 + hi * 8 + 16);
    v16bf pa1 = load_frag(P + (1 * 16 + rlo) * 32 + hi * 8,
                          P + (1 * 16 + rlo) * 32 + hi * 8 + 16);
    const int kb = kv0 + hi * 8;
#pragma unroll
    for (int ch = 0; ch < 4; ++ch) {
      const bf16* pv = Vb + (size_t)(ch * 16 + rlo) * TT + kb;
      v16bf vf = load_frag(pv, pv + 16);
      o[0][ch] = wmma_bf16(pa0, vf, o[0][ch]);
      o[1][ch] = wmma_bf16(pa1, vf, o[1][ch]);
    }
  }

  // normalize and write back to [B,T,C] bf16
  const int b = bh >> 4, h = bh & 15;
#pragma unroll
  for (int qi = 0; qi < 2; ++qi) {
#pragma unroll
    for (int r = 0; r < 8; ++r) {
      const int m = r + hi * 8;
      const int t = qbase + qi * 16 + m;
      const float inv = 1.0f / lR[qi][r];
      size_t rowoff = ((size_t)b * TT + t) * CC + h * DD;
      Y[rowoff + 0 * 16 + rlo] = (bf16)(o[qi][0][r] * inv);
      Y[rowoff + 1 * 16 + rlo] = (bf16)(o[qi][1][r] * inv);
      Y[rowoff + 2 * 16 + rlo] = (bf16)(o[qi][2][r] * inv);
      Y[rowoff + 3 * 16 + rlo] = (bf16)(o[qi][3][r] * inv);
    }
  }
}

// ---------------- output projection: [4096,1024] x [1024,1024] ----------------
// Same 32x64-per-wave WMMA tiling, f32 output + bias.

__global__ void proj_gemm(const bf16* __restrict__ Y, const bf16* __restrict__ WT,
                          const float* __restrict__ bias, float* __restrict__ out) {
  const int lane = threadIdx.x & 31;
  const int wid  = threadIdx.x >> 5;
  const int tile = blockIdx.x * 4 + wid;
  const int NT = CC / 64;                 // 16
  const int tm = tile / NT;
  const int tn = tile - tm * NT;
  const int rlo = lane & 15, hi = lane >> 4;

  const bf16* arow0 = Y + (size_t)(tm * 32 + rlo) * CC;
  const bf16* arow1 = Y + (size_t)(tm * 32 + 16 + rlo) * CC;
  const bf16* brow[4];
#pragma unroll
  for (int ni = 0; ni < 4; ++ni)
    brow[ni] = WT + (size_t)(tn * 64 + ni * 16 + rlo) * CC;

  v8f acc[2][4] = {};
#pragma unroll 2
  for (int kk = 0; kk < CC; kk += 32) {
    const int kb = kk + hi * 8;
    v16bf a0 = load_frag(arow0 + kb, arow0 + kb + 16);
    v16bf a1 = load_frag(arow1 + kb, arow1 + kb + 16);
    v16bf b0 = load_frag(brow[0] + kb, brow[0] + kb + 16);
    v16bf b1 = load_frag(brow[1] + kb, brow[1] + kb + 16);
    v16bf b2 = load_frag(brow[2] + kb, brow[2] + kb + 16);
    v16bf b3 = load_frag(brow[3] + kb, brow[3] + kb + 16);
    acc[0][0] = wmma_bf16(a0, b0, acc[0][0]);
    acc[0][1] = wmma_bf16(a0, b1, acc[0][1]);
    acc[0][2] = wmma_bf16(a0, b2, acc[0][2]);
    acc[0][3] = wmma_bf16(a0, b3, acc[0][3]);
    acc[1][0] = wmma_bf16(a1, b0, acc[1][0]);
    acc[1][1] = wmma_bf16(a1, b1, acc[1][1]);
    acc[1][2] = wmma_bf16(a1, b2, acc[1][2]);
    acc[1][3] = wmma_bf16(a1, b3, acc[1][3]);
  }

#pragma unroll
  for (int ni = 0; ni < 4; ++ni) {
    const int j = tn * 64 + ni * 16 + rlo;
    const float bj = bias[j];
#pragma unroll
    for (int mi = 0; mi < 2; ++mi) {
#pragma unroll
      for (int r = 0; r < 8; ++r) {
        const int i = tm * 32 + mi * 16 + r + hi * 8;
        out[(size_t)i * CC + j] = acc[mi][ni][r] + bj;
      }
    }
  }
}

// ---------------- host launch ----------------

extern "C" void kernel_launch(void* const* d_in, const int* in_sizes, int n_in,
                              void* d_out, int out_size, void* d_ws, size_t ws_size,
                              hipStream_t stream) {
  const float* x      = (const float*)d_in[0];
  const float* W_attn = (const float*)d_in[1];
  const float* b_attn = (const float*)d_in[2];
  const float* W_proj = (const float*)d_in[3];
  const float* b_proj = (const float*)d_in[4];
  float* out = (float*)d_out;

  char* ws = (char*)d_ws;
  const size_t MB = 1024 * 1024;
  bf16* Xbf = (bf16*)(ws + 0 * MB);    // [4096][1024]           8 MB
  bf16* WaT = (bf16*)(ws + 8 * MB);    // [3072][1024] = Wa^T    6 MB
  bf16* WpT = (bf16*)(ws + 14 * MB);   // [1024][1024] = Wp^T    2 MB
  bf16* Qb  = (bf16*)(ws + 16 * MB);   // [32][2048][64]         8 MB
  bf16* Kb  = (bf16*)(ws + 24 * MB);   // [32][2048][64]         8 MB
  bf16* Vt  = (bf16*)(ws + 32 * MB);   // [32][64][2048]         8 MB
  bf16* Yb  = (bf16*)(ws + 40 * MB);   // [4096][1024]           8 MB

  const int nX  = BN * TT * CC;        // 4,194,304
  const int nWa = CC * 3 * CC;         // 3,145,728
  const int nWp = CC * CC;             // 1,048,576

  cvt_f32_bf16<<<(nX + 255) / 256, 256, 0, stream>>>(x, Xbf, nX);
  transpose_f32_bf16<<<(nWa + 255) / 256, 256, 0, stream>>>(W_attn, WaT, CC, 3 * CC);
  transpose_f32_bf16<<<(nWp + 255) / 256, 256, 0, stream>>>(W_proj, WpT, CC, CC);

  // QKV: (4096/32)*(3072/64) = 6144 tiles, 4 waves/block
  qkv_gemm<<<6144 / 4, 128, 0, stream>>>(Xbf, WaT, b_attn, Qb, Kb, Vt);

  // attention: B*H*(T/32) = 2048 waves
  attn<<<2048 / 4, 128, 0, stream>>>(Qb, Kb, Vt, Yb);

  // projection: (4096/32)*(1024/64) = 2048 tiles
  proj_gemm<<<2048 / 4, 128, 0, stream>>>(Yb, WpT, b_proj, out);
}
